// pairWiseModel_54322746359987
// MI455X (gfx1250) — matrix operand order repaired and checked by reading, not compiled, not run on hardware
//
#include <hip/hip_runtime.h>
#include <hip/hip_bf16.h>
#include <math.h>

typedef __attribute__((ext_vector_type(16))) _Float16 v16h;
typedef __attribute__((ext_vector_type(8)))  float    v8f;

// ---------------- problem constants (from reference) ----------------
#define F_IN   136
#define FPAD   160      // F padded to multiple of 32 (5 k-blocks)
#define H1     256
#define H2     128
#define MT     128      // rows per workgroup (8 M-tiles, one per wave)

#define SA_STRIDE 168   // f16 stride for feature tile (mult of 8 for b128 loads)
#define H1S       264   // f16 stride for h1 tile
#define H2S       136   // f16 stride for h2 tile

#define KB1 5           // FPAD/32
#define KB2 8           // H1/32
#define NT1 16          // H1/16
#define NT2 8           // H2/16

// ws layout (bytes)
#define W1F_BYTES (NT1*KB1*32*16*2)   // 81920
#define W2F_BYTES (NT2*KB2*32*16*2)   // 65536
#define W1F_OFF   0
#define W2F_OFF   (W1F_OFF + W1F_BYTES)
#define SCORES_OFF (W2F_OFF + W2F_BYTES)

// dynamic LDS layout (bytes)
#define LDS_W1_OFF 0
#define LDS_W2_OFF (LDS_W1_OFF + W1F_BYTES)                  // 81920
#define LDS_A_OFF  (LDS_W2_OFF + W2F_BYTES)                  // 147456
#define LDS_A_BYTES (MT * SA_STRIDE * 2)                     // 43008
#define LDS_H1_OFF (LDS_A_OFF + LDS_A_BYTES)                 // 190464
#define LDS_H1_BYTES (MT * H1S * 2)                          // 67584
#define LDS_TOTAL  (LDS_H1_OFF + LDS_H1_BYTES)               // 258048 <= 320KB

#define LOSS_BLOCKS 1024

// ---------------- weight conversion: f32 -> f16 pre-swizzled B fragments ----
// B fragment lane layout (16-bit B, 32x16): lane L holds N = L&15,
// K-local = (L<16 ? 0 : 8) + (i<8 ? i : 8+i) for value i in 0..15.
__global__ __launch_bounds__(256) void convert_weights_kernel(
    const float* __restrict__ W1, const float* __restrict__ W2,
    _Float16* __restrict__ w1f, _Float16* __restrict__ w2f) {
  int t = blockIdx.x * 256 + threadIdx.x;
  if (t < NT1 * KB1 * 32) {
    int lane = t & 31;
    int rest = t >> 5;
    int kb = rest % KB1;
    int nTile = rest / KB1;
    int n = nTile * 16 + (lane & 15);
    int kOff = kb * 32 + ((lane & 16) ? 8 : 0);
    _Float16 vals[16];
#pragma unroll
    for (int i = 0; i < 16; ++i) {
      int k = kOff + (i < 8 ? i : 8 + i);
      vals[i] = (_Float16)((k < F_IN) ? W1[k * H1 + n] : 0.0f);
    }
    uint4* dst = (uint4*)(w1f + (size_t)t * 16);
    dst[0] = ((const uint4*)vals)[0];
    dst[1] = ((const uint4*)vals)[1];
  } else if (t < NT1 * KB1 * 32 + NT2 * KB2 * 32) {
    int t2 = t - NT1 * KB1 * 32;
    int lane = t2 & 31;
    int rest = t2 >> 5;
    int kb = rest & (KB2 - 1);
    int nTile = rest >> 3;
    int n = nTile * 16 + (lane & 15);
    int kOff = kb * 32 + ((lane & 16) ? 8 : 0);
    _Float16 vals[16];
#pragma unroll
    for (int i = 0; i < 16; ++i) {
      int k = kOff + (i < 8 ? i : 8 + i);
      vals[i] = (_Float16)W2[k * H2 + n];
    }
    uint4* dst = (uint4*)(w2f + (size_t)t2 * 16);
    dst[0] = ((const uint4*)vals)[0];
    dst[1] = ((const uint4*)vals)[1];
  }
}

// ---------------- fragment loaders ----------------
// A fragment (16x32 f16) from LDS, row-major [row][k], stride mult of 8.
__device__ __forceinline__ v16h load_frag_a(const _Float16* lds, int stride,
                                            int mBase, int kBase, int lane) {
  int m = mBase + (lane & 15);
  int k0 = kBase + ((lane >> 4) << 3);     // +0 (lanes 0-15) or +8 (lanes 16-31)
  const _Float16* p = lds + m * stride + k0;
  v16h out;
  ((uint4*)&out)[0] = *(const uint4*)(p);        // K = k0 .. k0+7   -> V0..V3
  ((uint4*)&out)[1] = *(const uint4*)(p + 16);   // K = k0+16..k0+23 -> V4..V7
  return out;
}

// B fragment from pre-swizzled buffer (LDS): 32 lanes x 32 bytes contiguous.
__device__ __forceinline__ v16h load_frag_b(const _Float16* frag, int tileIdx, int lane) {
  const uint4* p = (const uint4*)(frag + ((size_t)(tileIdx * 32 + lane)) * 16);
  v16h out;
  ((uint4*)&out)[0] = p[0];
  ((uint4*)&out)[1] = p[1];
  return out;
}

// ---------------- fused 3-layer MLP scorer (weights staged in LDS) ----------
__global__ __launch_bounds__(256) void mlp_score_kernel(
    const float* __restrict__ feat,
    const float* __restrict__ b1v, const float* __restrict__ b2v,
    const float* __restrict__ W3, const float* __restrict__ b3v,
    const _Float16* __restrict__ w1f, const _Float16* __restrict__ w2f,
    float* __restrict__ scores, int N) {
  extern __shared__ char smem[];
  _Float16* sW1 = (_Float16*)(smem + LDS_W1_OFF);   // 80 KB weight fragments
  _Float16* sW2 = (_Float16*)(smem + LDS_W2_OFF);   // 64 KB weight fragments
  _Float16* sA  = (_Float16*)(smem + LDS_A_OFF);    // feature tile, reused as h2
  _Float16* sH1 = (_Float16*)(smem + LDS_H1_OFF);   // h1 tile
  _Float16* sH2 = sA;                               // 128*136*2 = 34816 <= 43008

  const int tid  = threadIdx.x;
  const int lane = tid & 31;
  const int w    = tid >> 5;                // wave id 0..7 == M-tile id
  const long rowBase = (long)blockIdx.x * MT;

  // ---- preload biases into registers (removes per-tile global stalls) ----
  float bias1[NT1];
#pragma unroll
  for (int t = 0; t < NT1; ++t) bias1[t] = b1v[t * 16 + (lane & 15)];
  float bias2[NT2];
#pragma unroll
  for (int t = 0; t < NT2; ++t) bias2[t] = b2v[t * 16 + (lane & 15)];

  // ---- stage 0a: stage pre-swizzled weight fragments into LDS ----
  {
    const uint4* g1 = (const uint4*)w1f;    // 5120 uint4
    uint4* l1 = (uint4*)sW1;
    for (int i = tid; i < W1F_BYTES / 16; i += 256) l1[i] = g1[i];
    const uint4* g2 = (const uint4*)w2f;    // 4096 uint4
    uint4* l2 = (uint4*)sW2;
    for (int i = tid; i < W2F_BYTES / 16; i += 256) l2[i] = g2[i];
  }

  // ---- stage 0b: load features, convert to f16, zero-pad K to FPAD ----
  for (int idx = tid; idx < MT * (FPAD / 2); idx += 256) {
    int r = idx / (FPAD / 2);
    int k = (idx - r * (FPAD / 2)) * 2;     // even; F_IN even so pairs don't straddle
    long gr = rowBase + r;
    float2 v = make_float2(0.0f, 0.0f);
    if (k < F_IN && gr < N) v = *(const float2*)(feat + gr * F_IN + k);
    sA[r * SA_STRIDE + k]     = (_Float16)v.x;
    sA[r * SA_STRIDE + k + 1] = (_Float16)v.y;
  }
  __syncthreads();

  const int mOff = w * 16 + ((lane >> 4) << 3);
  const int nLo  = lane & 15;

  // ---- GEMM1: [128,160] x [160,256] -> h1, bias+ReLU, to LDS f16 ----
  // B fragments double-buffered: next tile's LDS loads overlap the WMMA chain.
  {
    v16h a[KB1];
#pragma unroll
    for (int kb = 0; kb < KB1; ++kb)
      a[kb] = load_frag_a(sA, SA_STRIDE, w * 16, kb * 32, lane);

    v16h bcur[KB1], bnxt[KB1];
#pragma unroll
    for (int kb = 0; kb < KB1; ++kb) bcur[kb] = load_frag_b(sW1, kb, lane);

#pragma unroll
    for (int nTile = 0; nTile < NT1; ++nTile) {
      if (nTile + 1 < NT1) {
#pragma unroll
        for (int kb = 0; kb < KB1; ++kb)
          bnxt[kb] = load_frag_b(sW1, (nTile + 1) * KB1 + kb, lane);
      }
      v8f acc = {0.f, 0.f, 0.f, 0.f, 0.f, 0.f, 0.f, 0.f};
#pragma unroll
      for (int kb = 0; kb < KB1; ++kb)
        acc = __builtin_amdgcn_wmma_f32_16x16x32_f16(
            false, a[kb], false, bcur[kb], (short)0, acc, false, false);

      int n = nTile * 16 + nLo;
#pragma unroll
      for (int r = 0; r < 8; ++r) {
        float vv = acc[r] + bias1[nTile];
        sH1[(mOff + r) * H1S + n] = (_Float16)(vv > 0.f ? vv : 0.f);
      }
#pragma unroll
      for (int kb = 0; kb < KB1; ++kb) bcur[kb] = bnxt[kb];
    }
  }
  __syncthreads();

  // ---- GEMM2: [128,256] x [256,128] -> h2, bias+ReLU, to LDS f16 ----
  // All K-block B loads grouped ahead of the WMMA chain (progressive waits).
  {
    v16h a[KB2];
#pragma unroll
    for (int kb = 0; kb < KB2; ++kb)
      a[kb] = load_frag_a(sH1, H1S, w * 16, kb * 32, lane);

#pragma unroll
    for (int nTile = 0; nTile < NT2; ++nTile) {
      v16h b[KB2];
#pragma unroll
      for (int kb = 0; kb < KB2; ++kb)
        b[kb] = load_frag_b(sW2, nTile * KB2 + kb, lane);

      v8f acc = {0.f, 0.f, 0.f, 0.f, 0.f, 0.f, 0.f, 0.f};
#pragma unroll
      for (int kb = 0; kb < KB2; ++kb)
        acc = __builtin_amdgcn_wmma_f32_16x16x32_f16(
            false, a[kb], false, b[kb], (short)0, acc, false, false);

      int n = nTile * 16 + nLo;
#pragma unroll
      for (int r = 0; r < 8; ++r) {
        float vv = acc[r] + bias2[nTile];
        sH2[(mOff + r) * H2S + n] = (_Float16)(vv > 0.f ? vv : 0.f);
      }
    }
  }
  __syncthreads();

  // ---- GEMM3: per-row dot(h2[row], W3) + b3 -> scores ----
  {
    int row  = tid >> 1;                    // 256 threads -> 128 rows, 2/row
    int part = tid & 1;
    float s = 0.f;
#pragma unroll
    for (int k = 0; k < 64; ++k) {
      int kk = part * 64 + k;
      s += (float)sH2[row * H2S + kk] * W3[kk];
    }
    s += __shfl_xor(s, 1, 32);              // partner shares the row
    long gr = rowBase + row;
    if (part == 0 && gr < N) scores[gr] = s + b3v[0];
  }
}

// ---------------- pairwise RankNet BCE: block partials ----------------
__global__ __launch_bounds__(256) void loss_partial_kernel(
    const float* __restrict__ scores, const int* __restrict__ target,
    const int* __restrict__ idx_i, const int* __restrict__ idx_j,
    int P, float* __restrict__ partials) {
  __shared__ float red[8];
  float acc = 0.f;
  for (long p = (long)blockIdx.x * 256 + threadIdx.x; p < P;
       p += (long)gridDim.x * 256) {
    int i = idx_i[p];
    int j = idx_j[p];
    float d = scores[i] - scores[j];            // SIGMA = 1.0
    int ti = target[i], tj = target[j];
    float S = (float)((ti > tj) - (ti < tj));   // sign in {-1,0,1}
    // log(sigmoid(d)) stably; log(1 - sigmoid(d)) = log(sigmoid(d)) - d
    float lp = (d >= 0.f) ? -log1pf(__expf(-d)) : (d - log1pf(__expf(d)));
    float lm = lp - d;
    lp = fmaxf(lp, -100.f);
    lm = fmaxf(lm, -100.f);
    acc -= S * lp + (1.f - S) * lm;
  }
#pragma unroll
  for (int o = 16; o > 0; o >>= 1) acc += __shfl_down(acc, o, 32);
  int lane = threadIdx.x & 31, wv = threadIdx.x >> 5;
  if (lane == 0) red[wv] = acc;
  __syncthreads();
  if (threadIdx.x == 0) {
    float s = 0.f;
#pragma unroll
    for (int k = 0; k < 8; ++k) s += red[k];
    partials[blockIdx.x] = s;
  }
}

// ---------------- deterministic final reduction ----------------
__global__ __launch_bounds__(256) void loss_final_kernel(
    const float* __restrict__ partials, int nPart, float* __restrict__ out) {
  __shared__ float red[8];
  float acc = 0.f;
  for (int p = threadIdx.x; p < nPart; p += 256) acc += partials[p];
#pragma unroll
  for (int o = 16; o > 0; o >>= 1) acc += __shfl_down(acc, o, 32);
  int lane = threadIdx.x & 31, wv = threadIdx.x >> 5;
  if (lane == 0) red[wv] = acc;
  __syncthreads();
  if (threadIdx.x == 0) {
    float s = 0.f;
#pragma unroll
    for (int k = 0; k < 8; ++k) s += red[k];
    out[0] = s;
  }
}

// ---------------- launcher ----------------
extern "C" void kernel_launch(void* const* d_in, const int* in_sizes, int n_in,
                              void* d_out, int out_size, void* d_ws, size_t ws_size,
                              hipStream_t stream) {
  const float* feat   = (const float*)d_in[0];
  const int*   target = (const int*)  d_in[1];
  const int*   idx_i  = (const int*)  d_in[2];
  const int*   idx_j  = (const int*)  d_in[3];
  const float* W1     = (const float*)d_in[4];
  const float* b1v    = (const float*)d_in[5];
  const float* W2     = (const float*)d_in[6];
  const float* b2v    = (const float*)d_in[7];
  const float* W3     = (const float*)d_in[8];
  const float* b3v    = (const float*)d_in[9];

  const int N = in_sizes[1];   // 500000
  const int P = in_sizes[2];   // 1000000

  char* ws = (char*)d_ws;
  _Float16* w1f    = (_Float16*)(ws + W1F_OFF);
  _Float16* w2f    = (_Float16*)(ws + W2F_OFF);
  float*    scores = (float*)  (ws + SCORES_OFF);
  size_t scoresBytes = ((size_t)N * 4 + 255) & ~(size_t)255;
  float*    partials = (float*)(ws + SCORES_OFF + scoresBytes);

  // 1) convert + pre-swizzle weights into WMMA B-fragment layout
  int convThreads = NT1 * KB1 * 32 + NT2 * KB2 * 32;   // 4608
  convert_weights_kernel<<<(convThreads + 255) / 256, 256, 0, stream>>>(
      W1, W2, w1f, w2f);

  // 2) fused 3-layer MLP, 128 rows per workgroup, weights + intermediates in
  //    252 KB of dynamic LDS (CDNA5 WGP has 320 KB)
  int nBlocks = (N + MT - 1) / MT;
  mlp_score_kernel<<<nBlocks, 256, LDS_TOTAL, stream>>>(
      feat, b1v, b2v, W3, b3v, w1f, w2f, scores, N);

  // 3) pairwise BCE, deterministic two-stage reduction
  loss_partial_kernel<<<LOSS_BLOCKS, 256, 0, stream>>>(
      scores, target, idx_i, idx_j, P, partials);
  loss_final_kernel<<<1, 256, 0, stream>>>(partials, LOSS_BLOCKS, (float*)d_out);
}
